// CrossAttention_67929202753593
// MI455X (gfx1250) — matrix operand order repaired and checked by reading, not compiled, not run on hardware
//
#include <hip/hip_runtime.h>
#include <hip/hip_bf16.h>

typedef __bf16 bf16_t;
typedef __attribute__((ext_vector_type(16))) __bf16 v16bf;
typedef __attribute__((ext_vector_type(8))) float v8f;

#define NUMH 8
#define SEQ  2048
#define BATCH 4
#define DHEAD 160   // 8 mv-channels * 16 components + 32 scalars

// ---------- small helpers ----------

__device__ inline int gradeOf(int x) {            // GRADE = [0,1,1,1,1,2,2,2,2,2,2,3,3,3,3,4]
  return (x == 0) ? 0 : (x < 5) ? 1 : (x < 11) ? 2 : (x < 15) ? 3 : 4;
}
__device__ inline float metricOf(int x) {         // METRIC = [+,+,-,-,-,-,-,-,+,+,+,+,+,+,-,-]
  return (x <= 1 || (x >= 8 && x <= 13)) ? 1.f : -1.f;
}
__device__ inline unsigned short f2bfu(float f) { // round-to-nearest-even fp32 -> bf16 bits
  unsigned u = __builtin_bit_cast(unsigned, f);
  return (unsigned short)((u + 0x7fffu + ((u >> 16) & 1u)) >> 16);
}
__device__ inline bf16_t f2bf(float f) {
  return __builtin_bit_cast(bf16_t, f2bfu(f));
}
// pack two floats as adjacent bf16 (RNE) in one dword via v_perm_b32
__device__ inline unsigned packbf2(float a, float b) {
  unsigned ua = __builtin_bit_cast(unsigned, a);
  unsigned ub = __builtin_bit_cast(unsigned, b);
  ua = ua + 0x7fffu + ((ua >> 16) & 1u);
  ub = ub + 0x7fffu + ((ub >> 16) & 1u);
  return __builtin_amdgcn_perm(ub, ua, 0x07060302u); // {ub[31:16], ua[31:16]}
}

struct U4x2 { uint4 a, b; };
struct U8 { unsigned u[8]; };

__device__ inline v16bf ld32(const bf16_t* p) {           // 32 contiguous bytes
  U4x2 t;
  t.a = *reinterpret_cast<const uint4*>(p);
  t.b = *reinterpret_cast<const uint4*>(p + 8);
  return __builtin_bit_cast(v16bf, t);
}
__device__ inline v16bf ld_pair(const bf16_t* p0, const bf16_t* p1) { // 2 x 16 bytes
  U4x2 t;
  t.a = *reinterpret_cast<const uint4*>(p0);
  t.b = *reinterpret_cast<const uint4*>(p1);
  return __builtin_bit_cast(v16bf, t);
}
__device__ inline v8f wmma_bf16(v16bf a, v16bf b, v8f c) {
  return __builtin_amdgcn_wmma_f32_16x16x32_bf16(false, a, false, b, (short)0, c, false, false);
}

// online-softmax update for one 16-row M-tile (two 16-key score fragments),
// rescales the O accumulators and writes packed bf16 P into this tile's LDS buffer
__device__ inline void softmax_update(const v8f& Sa, const v8f& Sb,
                                      float* mrun, float* lsum, v8f* O,
                                      unsigned* lbuf, int m, int half) {
  #pragma unroll
  for (int r = 0; r < 8; ++r) {
    float mx = fmaxf(Sa[r], Sb[r]);
    mx = fmaxf(mx, __shfl_xor(mx, 1));
    mx = fmaxf(mx, __shfl_xor(mx, 2));
    mx = fmaxf(mx, __shfl_xor(mx, 4));
    mx = fmaxf(mx, __shfl_xor(mx, 8));
    const float mnew  = fmaxf(mrun[r], mx);
    const float alpha = __expf(mrun[r] - mnew);
    mrun[r] = mnew;
    const float p0 = __expf(Sa[r] - mnew);
    const float p1 = __expf(Sb[r] - mnew);
    float rs = p0 + p1;
    rs += __shfl_xor(rs, 1);
    rs += __shfl_xor(rs, 2);
    rs += __shfl_xor(rs, 4);
    rs += __shfl_xor(rs, 8);
    lsum[r] = lsum[r] * alpha + rs;
    #pragma unroll
    for (int c = 0; c < 10; ++c) O[c][r] *= alpha;
    lbuf[(r + half * 8) * 20 + m] = packbf2(p0, p1);
  }
}

// rebuild a 16x32 P A-fragment from the packed LDS tile: lane m = query row
__device__ inline v16bf read_pfrag(const unsigned* lbuf, int m, int half) {
  unsigned w[8];
  #pragma unroll
  for (int j = 0; j < 8; ++j) w[j] = lbuf[m * 20 + half * 8 + j];
  U8 t;
  #pragma unroll
  for (int j = 0; j < 4; ++j) {
    t.u[j]     = __builtin_amdgcn_perm(w[2 * j + 1], w[2 * j], 0x05040100u); // low halves
    t.u[j + 4] = __builtin_amdgcn_perm(w[2 * j + 1], w[2 * j], 0x07060302u); // high halves
  }
  return __builtin_bit_cast(v16bf, t);
}

// ---------- kernel 1: Q projection -> Qb[b][h][n][160] bf16, scale folded in ----------

__global__ __launch_bounds__(256) void qproj_kernel(
    const float* __restrict__ mvq, const float* __restrict__ sq,
    const float* __restrict__ w_mv, const float* __restrict__ w_s2mv,
    const float* __restrict__ w_mv2s, const float* __restrict__ w_s,
    const float* __restrict__ b_mv, const float* __restrict__ b_s,
    bf16_t* __restrict__ Qb)
{
  __shared__ float smv[4][256];
  __shared__ float ssc[4][64];
  const int t = threadIdx.x;
  const int tok0 = blockIdx.x * 4;
  for (int idx = t; idx < 1024; idx += 256)
    smv[idx >> 8][idx & 255] = mvq[(size_t)tok0 * 256 + idx];
  if (t < 256) ssc[t >> 6][t & 63] = sq[(size_t)tok0 * 64 + t];
  __syncthreads();
  const int tt = t >> 6, lt = t & 63;
  const int tok = tok0 + tt;
  const int b = tok >> 11, n = tok & 2047;
  const float* MV = smv[tt];
  const float* SS = ssc[tt];
  const float scale = 0.07905694150420949f;  // 1/sqrt(160)

  { // multivector channel o = lt (0..63): out[o][x] = sum_i w[o,i,GRADE[x]]*mv[i,x]
    const int o = lt;
    float acc[16];
    #pragma unroll
    for (int x = 0; x < 16; ++x) acc[x] = 0.f;
    for (int i = 0; i < 16; ++i) {
      const float* wr = w_mv + (size_t)(o * 16 + i) * 5;
      float wg[5] = {wr[0], wr[1], wr[2], wr[3], wr[4]};
      const float* mvi = MV + i * 16;
      #pragma unroll
      for (int x = 0; x < 16; ++x) acc[x] = fmaf(wg[gradeOf(x)], mvi[x], acc[x]);
    }
    float a0 = b_mv[o];
    for (int j = 0; j < 64; ++j) a0 = fmaf(w_s2mv[o * 64 + j], SS[j], a0);
    acc[0] += a0;
    const int c = o >> 3, h = o & 7;           // reshape (Cm,H): o = c*8 + h
    bf16_t* dst = Qb + (((size_t)(b * NUMH + h) * SEQ + n) * DHEAD + c * 16);
    #pragma unroll
    for (int x = 0; x < 16; ++x) dst[x] = f2bf(acc[x] * scale);
  }
  #pragma unroll
  for (int u = 0; u < 4; ++u) {               // scalar channels, 4 per thread (0..255)
    const int os = lt * 4 + u;
    float a = b_s[os];
    for (int j = 0; j < 64; ++j) a = fmaf(w_s[os * 64 + j], SS[j], a);
    for (int i = 0; i < 16; ++i) a = fmaf(w_mv2s[os * 16 + i], MV[i * 16], a);
    const int cs = os >> 3, h = os & 7;        // reshape (Cs,H): os = cs*8 + h
    Qb[((size_t)(b * NUMH + h) * SEQ + n) * DHEAD + 128 + cs] = f2bf(a * scale);
  }
}

// ---------- kernel 2: KV projection -> Kb[b][n][160] (METRIC folded), Vt[b][160][n] ----------

__global__ __launch_bounds__(256) void kvproj_kernel(
    const float* __restrict__ mvkv, const float* __restrict__ skv,
    const float* __restrict__ w_mv, const float* __restrict__ w_s2mv,
    const float* __restrict__ w_mv2s, const float* __restrict__ w_s,
    const float* __restrict__ b_mv, const float* __restrict__ b_s,
    bf16_t* __restrict__ Kb, bf16_t* __restrict__ Vt)
{
  __shared__ float smv[4][256];
  __shared__ float ssc[4][64];
  const int t = threadIdx.x;
  const int tok0 = blockIdx.x * 4;
  for (int idx = t; idx < 1024; idx += 256)
    smv[idx >> 8][idx & 255] = mvkv[(size_t)tok0 * 256 + idx];
  if (t < 256) ssc[t >> 6][t & 63] = skv[(size_t)tok0 * 64 + t];
  __syncthreads();
  const int tt = t >> 6, lt = t & 63;
  const int tok = tok0 + tt;
  const int b = tok >> 11, n = tok & 2047;
  const float* MV = smv[tt];
  const float* SS = ssc[tt];

  if (lt < 16) {                               // 16 mv channels: 0..7 = K, 8..15 = V
    const int o = lt;
    float acc[16];
    #pragma unroll
    for (int x = 0; x < 16; ++x) acc[x] = 0.f;
    for (int i = 0; i < 16; ++i) {
      const float* wr = w_mv + (size_t)(o * 16 + i) * 5;
      float wg[5] = {wr[0], wr[1], wr[2], wr[3], wr[4]};
      const float* mvi = MV + i * 16;
      #pragma unroll
      for (int x = 0; x < 16; ++x) acc[x] = fmaf(wg[gradeOf(x)], mvi[x], acc[x]);
    }
    float a0 = b_mv[o];
    for (int j = 0; j < 64; ++j) a0 = fmaf(w_s2mv[o * 64 + j], SS[j], a0);
    acc[0] += a0;
    if (o < 8) {
      bf16_t* dst = Kb + ((size_t)b * SEQ + n) * DHEAD + o * 16;
      #pragma unroll
      for (int x = 0; x < 16; ++x) dst[x] = f2bf(acc[x] * metricOf(x));
    } else {
      const int c = o - 8;
      #pragma unroll
      for (int x = 0; x < 16; ++x)
        Vt[(size_t)b * DHEAD * SEQ + (size_t)(c * 16 + x) * SEQ + n] = f2bf(acc[x]);
    }
  }
  {                                            // 64 scalar channels: 0..31 = K, 32..63 = V
    const int os = lt;
    float a = b_s[os];
    for (int j = 0; j < 64; ++j) a = fmaf(w_s[os * 64 + j], SS[j], a);
    for (int i = 0; i < 16; ++i) a = fmaf(w_mv2s[os * 16 + i], MV[i * 16], a);
    if (os < 32)
      Kb[((size_t)b * SEQ + n) * DHEAD + 128 + os] = f2bf(a);
    else
      Vt[(size_t)b * DHEAD * SEQ + (size_t)(128 + os - 32) * SEQ + n] = f2bf(a);
  }
}

// ---------- kernel 3: flash attention, one wave per 32-query tile ----------
// All K and V fragments for an iteration are loaded up-front: K loads complete
// before the score WMMAs; V loads drain under the score WMMAs + softmax VALU,
// so the PV WMMAs issue without vmem waits. Each B-fragment feeds two WMMAs.

__global__ __launch_bounds__(128) void attn_kernel(
    const bf16_t* __restrict__ Q, const bf16_t* __restrict__ K,
    const bf16_t* __restrict__ Vt, const float* __restrict__ head_scale,
    float* __restrict__ Hb)
{
  __shared__ unsigned pbuf[4][2][320];         // per-wave: two 16x16 P tiles, rows padded to 20 words
  const int lid  = threadIdx.x & 31;
  const int wave = threadIdx.x >> 5;
  const int g  = blockIdx.x * 4 + wave;        // 0..2047
  const int qt = g & (SEQ / 32 - 1);           // 64 tiles of 32 queries
  const int bh = g >> 6;
  const int h  = bh & (NUMH - 1);
  const int b  = bh >> 3;
  const bf16_t* Qp = Q  + ((size_t)(b * NUMH + h) * SEQ + qt * 32) * DHEAD;
  const bf16_t* Kp = K  + (size_t)b * SEQ * DHEAD;
  const bf16_t* Vp = Vt + (size_t)b * DHEAD * SEQ;
  float*        Hp = Hb + ((size_t)(b * NUMH + h) * SEQ + qt * 32) * DHEAD;

  const int m = lid & 15;
  const int half = lid >> 4;

  // Q as 2 x five 16x32 A-fragments: lane m = row; K-halves 0-7 / 8-15 (+16) per lane half
  v16bf aq0[5], aq1[5];
  const bf16_t* qrow0 = Qp + (size_t)m * DHEAD;
  const bf16_t* qrow1 = Qp + (size_t)(16 + m) * DHEAD;
  #pragma unroll
  for (int c = 0; c < 5; ++c) {
    aq0[c] = ld_pair(qrow0 + c * 32 + half * 8, qrow0 + c * 32 + 16 + half * 8);
    aq1[c] = ld_pair(qrow1 + c * 32 + half * 8, qrow1 + c * 32 + 16 + half * 8);
  }

  const v8f vzero = {0.f, 0.f, 0.f, 0.f, 0.f, 0.f, 0.f, 0.f};
  v8f O0[10], O1[10];
  #pragma unroll
  for (int c = 0; c < 10; ++c) { O0[c] = vzero; O1[c] = vzero; }
  float mrun0[8], lsum0[8], mrun1[8], lsum1[8];
  #pragma unroll
  for (int r = 0; r < 8; ++r) {
    mrun0[r] = -1e30f; lsum0[r] = 0.f;
    mrun1[r] = -1e30f; lsum1[r] = 0.f;
  }

  unsigned* lb0 = &pbuf[wave][0][0];
  unsigned* lb1 = &pbuf[wave][1][0];

  for (int kb = 0; kb < SEQ; kb += 32) {
    const bf16_t* kr0 = Kp + (size_t)(kb + m) * DHEAD;
    const bf16_t* kr1 = kr0 + 16 * DHEAD;
    // 1) issue all K fragment loads for this iteration
    v16bf kf0[5], kf1[5];
    #pragma unroll
    for (int c = 0; c < 5; ++c) {
      kf0[c] = ld_pair(kr0 + c * 32 + half * 8, kr0 + c * 32 + 16 + half * 8);
      kf1[c] = ld_pair(kr1 + c * 32 + half * 8, kr1 + c * 32 + 16 + half * 8);
    }
    // 2) issue all V fragment loads (in-order returns: drain after K, used after softmax)
    v16bf vf[10];
    #pragma unroll
    for (int c = 0; c < 10; ++c)
      vf[c] = ld_pair(Vp + (size_t)(c * 16 + m) * SEQ + kb + half * 8, Vp + (size_t)(c * 16 + m) * SEQ + kb + 16 + half * 8);
    // 3) prefetch next key block rows (L2 stream)
    if (kb + 32 < SEQ) {
      __builtin_prefetch(kr0 + 32 * DHEAD, 0, 3);
      __builtin_prefetch(kr1 + 32 * DHEAD, 0, 3);
    }
    // 4) scores: 2 M-tiles x 2 K-tiles, back-to-back WMMAs
    v8f S00 = vzero, S01 = vzero, S10 = vzero, S11 = vzero;
    #pragma unroll
    for (int c = 0; c < 5; ++c) {
      S00 = wmma_bf16(aq0[c], kf0[c], S00);
      S10 = wmma_bf16(aq1[c], kf0[c], S10);
      S01 = wmma_bf16(aq0[c], kf1[c], S01);
      S11 = wmma_bf16(aq1[c], kf1[c], S11);
    }
    // 5) online softmax per M-tile; P tiles go to disjoint LDS buffers
    softmax_update(S00, S01, mrun0, lsum0, O0, lb0, m, half);
    softmax_update(S10, S11, mrun1, lsum1, O1, lb1, m, half);
    asm volatile("s_wait_dscnt 0" ::: "memory");
    v16bf ap0 = read_pfrag(lb0, m, half);
    v16bf ap1 = read_pfrag(lb1, m, half);
    // 6) O += P x V; V fragments already resident
    #pragma unroll
    for (int c = 0; c < 10; ++c) {
      O0[c] = wmma_bf16(ap0, vf[c], O0[c]);
      O1[c] = wmma_bf16(ap1, vf[c], O1[c]);
    }
  }
  const float hs = head_scale[h];
  float invl0[8], invl1[8];
  #pragma unroll
  for (int r = 0; r < 8; ++r) {
    invl0[r] = hs / lsum0[r];
    invl1[r] = hs / lsum1[r];
  }
  #pragma unroll
  for (int c = 0; c < 10; ++c)
    #pragma unroll
    for (int r = 0; r < 8; ++r) {
      Hp[(size_t)(r + half * 8) * DHEAD + c * 16 + m]        = O0[c][r] * invl0[r];
      Hp[(size_t)(16 + r + half * 8) * DHEAD + c * 16 + m]   = O1[c][r] * invl1[r];
    }
}

// ---------- kernel 4: output projection ----------

__global__ __launch_bounds__(256) void oproj_kernel(
    const float* __restrict__ Hb,
    const float* __restrict__ w_mv, const float* __restrict__ w_s2mv,
    const float* __restrict__ w_mv2s, const float* __restrict__ w_s,
    const float* __restrict__ b_mv, const float* __restrict__ b_s,
    float* __restrict__ out_mv, float* __restrict__ out_s)
{
  __shared__ float hin[4][1280];               // [i*16+x] i<64 (i=h*8+c) ; [1024+j] j=h*32+cs
  const int t = threadIdx.x;
  const int tok0 = blockIdx.x * 4;
  for (int idx = t; idx < 4 * 1280; idx += 256) {
    const int tt = idx / 1280;
    const int f  = idx - tt * 1280;
    const int tok = tok0 + tt;
    const int b = tok >> 11, n = tok & 2047;
    float v;
    if (f < 1024) {
      const int i = f >> 4, x = f & 15;
      const int h = i >> 3, c = i & 7;
      v = Hb[((size_t)(b * NUMH + h) * SEQ + n) * DHEAD + c * 16 + x];
    } else {
      const int j = f - 1024;
      const int h = j >> 5, cs = j & 31;
      v = Hb[((size_t)(b * NUMH + h) * SEQ + n) * DHEAD + 128 + cs];
    }
    hin[tt][f] = v;
  }
  __syncthreads();
  const int tt = t >> 6, lt = t & 63;
  const int tok = tok0 + tt;
  const int b = tok >> 11, n = tok & 2047;
  const float* HI = hin[tt];
  {
    const int o  = lt >> 2;
    const int xb = (lt & 3) * 4;
    float acc[4] = {0.f, 0.f, 0.f, 0.f};
    for (int i = 0; i < 64; ++i) {
      const float* wr = w_mv + (size_t)(o * 64 + i) * 5;
      const float* hi = HI + i * 16 + xb;
      #pragma unroll
      for (int u = 0; u < 4; ++u) acc[u] = fmaf(wr[gradeOf(xb + u)], hi[u], acc[u]);
    }
    if (xb == 0) {
      float a0 = b_mv[o];
      for (int j = 0; j < 256; ++j) a0 = fmaf(w_s2mv[o * 256 + j], HI[1024 + j], a0);
      acc[0] += a0;
    }
    float* dst = out_mv + (((size_t)b * SEQ + n) * 16 + o) * 16 + xb;
    #pragma unroll
    for (int u = 0; u < 4; ++u) dst[u] = acc[u];
  }
  {
    const int os = lt;
    float a = b_s[os];
    for (int j = 0; j < 256; ++j) a = fmaf(w_s[os * 256 + j], HI[1024 + j], a);
    for (int i = 0; i < 64; ++i) a = fmaf(w_mv2s[os * 64 + i], HI[i * 16], a);
    out_s[((size_t)b * SEQ + n) * 64 + os] = a;
  }
}

// ---------- launch ----------

extern "C" void kernel_launch(void* const* d_in, const int* in_sizes, int n_in,
                              void* d_out, int out_size, void* d_ws, size_t ws_size,
                              hipStream_t stream) {
  (void)in_sizes; (void)n_in; (void)out_size; (void)ws_size;
  const float* mv_kv   = (const float*)d_in[0];
  const float* mv_q    = (const float*)d_in[1];
  const float* s_kv    = (const float*)d_in[2];
  const float* s_q     = (const float*)d_in[3];
  const float* wq_mv   = (const float*)d_in[4];
  const float* wq_s2mv = (const float*)d_in[5];
  const float* wq_mv2s = (const float*)d_in[6];
  const float* wq_s    = (const float*)d_in[7];
  const float* bq_mv   = (const float*)d_in[8];
  const float* bq_s    = (const float*)d_in[9];
  const float* wkv_mv   = (const float*)d_in[10];
  const float* wkv_s2mv = (const float*)d_in[11];
  const float* wkv_mv2s = (const float*)d_in[12];
  const float* wkv_s    = (const float*)d_in[13];
  const float* bkv_mv   = (const float*)d_in[14];
  const float* bkv_s    = (const float*)d_in[15];
  const float* wo_mv   = (const float*)d_in[16];
  const float* wo_s2mv = (const float*)d_in[17];
  const float* wo_mv2s = (const float*)d_in[18];
  const float* wo_s    = (const float*)d_in[19];
  const float* bo_mv   = (const float*)d_in[20];
  const float* bo_s    = (const float*)d_in[21];
  const float* head_scale = (const float*)d_in[22];

  char* ws = (char*)d_ws;
  const size_t qb_bytes = (size_t)BATCH * NUMH * SEQ * DHEAD * 2;  // 20 MB
  const size_t kb_bytes = (size_t)BATCH * SEQ * DHEAD * 2;         // 2.6 MB
  bf16_t* Qb = (bf16_t*)ws;
  bf16_t* Kb = (bf16_t*)(ws + qb_bytes);
  bf16_t* Vt = (bf16_t*)(ws + qb_bytes + kb_bytes);
  float*  Hb = (float*) (ws + qb_bytes + 2 * kb_bytes);            // 42 MB

  float* out_mv = (float*)d_out;
  float* out_s  = out_mv + (size_t)BATCH * SEQ * 256;

  qproj_kernel<<<(BATCH * SEQ) / 4, 256, 0, stream>>>(
      mv_q, s_q, wq_mv, wq_s2mv, wq_mv2s, wq_s, bq_mv, bq_s, Qb);
  kvproj_kernel<<<(BATCH * SEQ) / 4, 256, 0, stream>>>(
      mv_kv, s_kv, wkv_mv, wkv_s2mv, wkv_mv2s, wkv_s, bkv_mv, bkv_s, Kb, Vt);
  attn_kernel<<<(BATCH * NUMH * (SEQ / 32)) / 4, 128, 0, stream>>>(
      Qb, Kb, Vt, head_scale, Hb);
  oproj_kernel<<<(BATCH * SEQ) / 4, 256, 0, stream>>>(
      Hb, wo_mv, wo_s2mv, wo_mv2s, wo_s, bo_mv, bo_s, out_mv, out_s);
}
